// AKDN_71287867179280
// MI455X (gfx1250) — matrix operand order repaired and checked by reading, not compiled, not run on hardware
//
#include <hip/hip_runtime.h>
#include <hip/hip_bf16.h>
#include <math.h>

typedef __attribute__((ext_vector_type(2)))  _Float16 v2h;
typedef __attribute__((ext_vector_type(8)))  _Float16 v8h;
typedef __attribute__((ext_vector_type(16))) _Float16 v16h;
typedef __attribute__((ext_vector_type(8)))  float    v8f;

#define KN_ENT 100000
#define KN_USR 50000
#define KD     64
#define WKT_S  136   // padded row stride (halves) for transposed Wk  (128 + 8)
#define WAT_S  72    // padded row stride (halves) for transposed Wa/Wb (64 + 8)

union V16U { v16h v; v8h h[2]; };

__device__ __forceinline__ v2h pk2(float x, float y) {
    return (v2h)__builtin_amdgcn_cvt_pkrtz(x, y);
}

// load 8 consecutive f32 from 16B-aligned p, convert to 8 packed halves
__device__ __forceinline__ v8h cvt8(const float* __restrict__ p) {
    float4 a = *(const float4*)p;
    float4 b = *(const float4*)(p + 4);
    union { v8h v; v2h p[4]; } u;
    u.p[0] = pk2(a.x, a.y);
    u.p[1] = pk2(a.z, a.w);
    u.p[2] = pk2(b.x, b.y);
    u.p[3] = pk2(b.z, b.w);
    return u.v;
}

__device__ __forceinline__ void atomicMaxF(float* addr, float val) {
    if (val >= 0.0f) atomicMax((int*)addr, __float_as_int(val));
    else             atomicMin((unsigned int*)addr, __float_as_uint(val));
}

__global__ void k_fill(float* __restrict__ p, float v, long n) {
    long i = (long)blockIdx.x * 256 + threadIdx.x;
    if (i < n) p[i] = v;
}

__global__ void k_add(float* __restrict__ y, const float* __restrict__ x, long n) {
    long i = (long)blockIdx.x * 256 + threadIdx.x;
    if (i < n) y[i] += x[i];
}

// ---------------------------------------------------------------------------
// Kernel 1: edge attention logits via WMMA
//   trans = [t_e ; h_e] (16x128) @ Wk (128x64) + b ; logit = <trans, r_e>
//   One wave = 16 edges; Wk kept transposed in LDS so B fragments are
//   two 8-half runs 16 apart along k (2x ds_load_b128 per fragment).
// ---------------------------------------------------------------------------
__global__ void __launch_bounds__(256)
k_logits(const float* __restrict__ emb, const float* __restrict__ rel,
         const float* __restrict__ Wk, const float* __restrict__ Wkb,
         const int* __restrict__ kg_h, const int* __restrict__ kg_t,
         const int* __restrict__ kg_r,
         float* __restrict__ logits, float* __restrict__ row_max, int nE)
{
    __shared__ _Float16 sWkT[64 * WKT_S];    // ~17 KB, transposed [n][k]
    __shared__ _Float16 sA[8][16 * 128];     // 32 KB, per-wave 16x128 A tile
    const int tid  = threadIdx.x;
    const int lane = tid & 31;
    const int wave = tid >> 5;
    const int hi   = lane >> 4;
    const int lo   = lane & 15;

    // stage Wk transposed: sWkT[n*WKT_S + k] = Wk[k*64 + n], packed pairs along k
    for (int i = tid; i < 64 * 64; i += 256) {          // 4096 half2 pairs
        const int n  = i & 63;
        const int k2 = (i >> 6) * 2;                    // 0,2,...,126
        *(v2h*)&sWkT[n * WKT_S + k2] = pk2(Wk[k2 * 64 + n], Wk[(k2 + 1) * 64 + n]);
    }

    // stage 16 edge rows: lo-half lanes fetch t_e, hi-half lanes fetch h_e
    const int base = (blockIdx.x * 8 + wave) * 16;
    for (int m = 0; m < 16; ++m) {
        int e = base + m; if (e >= nE) e = nE - 1;
        const int r = hi ? kg_h[e] : kg_t[e];
        float4 v = *(const float4*)&emb[(long)r * KD + lo * 4];
        union { uint2 u; v2h h[2]; } s;
        s.h[0] = pk2(v.x, v.y);
        s.h[1] = pk2(v.z, v.w);
        *(uint2*)&sA[wave][m * 128 + hi * 64 + lo * 4] = s.u;   // [0:64)=t, [64:128)=h
    }
    __syncthreads();

    int ridx[8];
    #pragma unroll
    for (int j = 0; j < 8; ++j) {
        int e = base + j + hi * 8; if (e >= nE) e = nE - 1;
        ridx[j] = kg_r[e];
    }

    float red[8];
    #pragma unroll
    for (int j = 0; j < 8; ++j) red[j] = 0.0f;

    for (int nt = 0; nt < 4; ++nt) {
        const int col = nt * 16 + lo;
        v8f acc = {};
        #pragma unroll
        for (int kc = 0; kc < 4; ++kc) {
            V16U a, b;
            const _Float16* ap = &sA[wave][lo * 128 + kc * 32 + hi * 8];
            a.h[0] = *(const v8h*)(ap);
            a.h[1] = *(const v8h*)(ap + 16);
            const _Float16* bp = &sWkT[col * WKT_S + kc * 32 + hi * 8];
            b.h[0] = *(const v8h*)(bp);
            b.h[1] = *(const v8h*)(bp + 16);
            acc = __builtin_amdgcn_wmma_f32_16x16x32_f16(false, a.v, false, b.v,
                                                         (short)0, acc, false, false);
        }
        const float bias = Wkb[col];
        #pragma unroll
        for (int j = 0; j < 8; ++j) {
            const float rv = rel[ridx[j] * KD + col];
            red[j] += (acc[j] + bias) * rv;
        }
    }

    #pragma unroll
    for (int j = 0; j < 8; ++j) {
        float r = red[j];
        r += __shfl_xor(r, 1, 32);
        r += __shfl_xor(r, 2, 32);
        r += __shfl_xor(r, 4, 32);
        r += __shfl_xor(r, 8, 32);
        red[j] = r;
    }

    if (lo == 0) {
        #pragma unroll
        for (int j = 0; j < 8; ++j) {
            const int e = base + j + hi * 8;
            if (e < nE) {
                float l = red[j];
                l = (l >= 0.0f) ? l : 0.01f * l;
                logits[e] = l;
                atomicMaxF(&row_max[kg_h[e]], l);
            }
        }
    }
}

// ---------------------------------------------------------------------------
// segment softmax passes
// ---------------------------------------------------------------------------
__global__ void k_exp(float* __restrict__ lg, const int* __restrict__ kg_h,
                      const float* __restrict__ rmax, float* __restrict__ rsum, int n) {
    int e = blockIdx.x * 256 + threadIdx.x;
    if (e >= n) return;
    int h = kg_h[e];
    float v = __expf(lg[e] - rmax[h]);
    lg[e] = v;
    atomicAdd(&rsum[h], v);
}

__global__ void k_norm(float* __restrict__ at, const int* __restrict__ kg_h,
                       const float* __restrict__ rsum, int n) {
    int e = blockIdx.x * 256 + threadIdx.x;
    if (e >= n) return;
    at[e] = at[e] / rsum[kg_h[e]];
}

// ---------------------------------------------------------------------------
// SpMM scatters: one thread = (edge, 4 columns); float4 gather + 4 f32 atomics
// ---------------------------------------------------------------------------
__global__ void k_spmm_kg(const float* __restrict__ attn, const int* __restrict__ kg_h,
                          const int* __restrict__ kg_t, const float* __restrict__ x,
                          float* __restrict__ out, int n) {
    long t = (long)blockIdx.x * 256 + threadIdx.x;
    long e = t >> 4;
    if (e >= n) return;
    int c = (int)(t & 15) * 4;
    float a = attn[e];
    long rt = kg_t[e], rh = kg_h[e];
    float4 v = *(const float4*)&x[rt * KD + c];
    float* o = &out[rh * KD + c];
    atomicAdd(o + 0, a * v.x);
    atomicAdd(o + 1, a * v.y);
    atomicAdd(o + 2, a * v.z);
    atomicAdd(o + 3, a * v.w);
}

__global__ void k_spmm_ain(const float* __restrict__ val, const int* __restrict__ rows,
                           const int* __restrict__ cols, const float* __restrict__ dual,
                           const float* __restrict__ users, float* __restrict__ out, int n) {
    long t = (long)blockIdx.x * 256 + threadIdx.x;
    long e = t >> 4;
    if (e >= n) return;
    int c = (int)(t & 15) * 4;
    float a = val[e];
    int cc = cols[e];
    long rr = rows[e];
    const float* src = (cc < KN_ENT) ? &dual[(long)cc * KD] : &users[(long)(cc - KN_ENT) * KD];
    float4 v = *(const float4*)&src[c];
    float* o = &out[rr * KD + c];
    atomicAdd(o + 0, a * v.x);
    atomicAdd(o + 1, a * v.y);
    atomicAdd(o + 2, a * v.z);
    atomicAdd(o + 3, a * v.w);
}

// ---------------------------------------------------------------------------
// fusion gate: g = sigmoid(kg@Wa + cb@Wb) ; dual = g*kg + (1-g)*cb   (WMMA)
// Wa/Wb transposed in LDS; row tiles staged packed.
// ---------------------------------------------------------------------------
__global__ void __launch_bounds__(256)
k_fusion(const float* __restrict__ kg, const float* __restrict__ cb,
         const float* __restrict__ Wa, const float* __restrict__ Wb,
         float* __restrict__ dual, int nRows)
{
    __shared__ _Float16 sWaT[64 * WAT_S];    // 9 KB, transposed [n][k]
    __shared__ _Float16 sWbT[64 * WAT_S];    // 9 KB
    __shared__ _Float16 sKG[8][16 * 64];     // 16 KB
    __shared__ _Float16 sCB[8][16 * 64];     // 16 KB
    const int tid = threadIdx.x, lane = tid & 31, wave = tid >> 5;
    const int hi = lane >> 4, lo = lane & 15;

    for (int i = tid; i < 64 * 32; i += 256) {       // 2048 half2 pairs each
        const int n  = i & 63;
        const int k2 = (i >> 6) * 2;                 // 0,2,...,62
        *(v2h*)&sWaT[n * WAT_S + k2] = pk2(Wa[k2 * 64 + n], Wa[(k2 + 1) * 64 + n]);
        *(v2h*)&sWbT[n * WAT_S + k2] = pk2(Wb[k2 * 64 + n], Wb[(k2 + 1) * 64 + n]);
    }

    const int base = (blockIdx.x * 8 + wave) * 16;
    for (int m = 0; m < 16; ++m) {
        int r = base + m; if (r >= nRows) r = nRows - 1;
        const float* srcp = hi ? &cb[(long)r * KD] : &kg[(long)r * KD];
        float4 v = *(const float4*)&srcp[lo * 4];
        union { uint2 u; v2h h[2]; } s;
        s.h[0] = pk2(v.x, v.y);
        s.h[1] = pk2(v.z, v.w);
        _Float16* dst = hi ? &sCB[wave][m * 64 + lo * 4] : &sKG[wave][m * 64 + lo * 4];
        *(uint2*)dst = s.u;
    }
    __syncthreads();

    for (int nt = 0; nt < 4; ++nt) {
        const int col = nt * 16 + lo;
        v8f acc = {};
        #pragma unroll
        for (int kc = 0; kc < 2; ++kc) {
            V16U a1, b1, a2, b2;
            const _Float16* a1p = &sKG[wave][lo * 64 + kc * 32 + hi * 8];
            const _Float16* a2p = &sCB[wave][lo * 64 + kc * 32 + hi * 8];
            a1.h[0] = *(const v8h*)(a1p);
            a1.h[1] = *(const v8h*)(a1p + 16);
            a2.h[0] = *(const v8h*)(a2p);
            a2.h[1] = *(const v8h*)(a2p + 16);
            const _Float16* b1p = &sWaT[col * WAT_S + kc * 32 + hi * 8];
            const _Float16* b2p = &sWbT[col * WAT_S + kc * 32 + hi * 8];
            b1.h[0] = *(const v8h*)(b1p);
            b1.h[1] = *(const v8h*)(b1p + 16);
            b2.h[0] = *(const v8h*)(b2p);
            b2.h[1] = *(const v8h*)(b2p + 16);
            acc = __builtin_amdgcn_wmma_f32_16x16x32_f16(false, a1.v, false, b1.v,
                                                         (short)0, acc, false, false);
            acc = __builtin_amdgcn_wmma_f32_16x16x32_f16(false, a2.v, false, b2.v,
                                                         (short)0, acc, false, false);
        }
        #pragma unroll
        for (int j = 0; j < 8; ++j) {
            const int m = j + hi * 8;
            const int r = base + m;
            if (r < nRows) {
                float g  = 1.0f / (1.0f + __expf(-acc[j]));
                float kv = (float)sKG[wave][m * 64 + col];
                float cv = (float)sCB[wave][m * 64 + col];
                dual[(long)r * KD + col] = g * kv + (1.0f - g) * cv;
            }
        }
    }
}

// ---------------------------------------------------------------------------
// final scores: out[u,i] = <user_sum[uid[u]-N_ENT], item_sum[iid[i]]>  (WMMA)
// fragments built from global f32 with float4 loads + packed converts
// ---------------------------------------------------------------------------
__global__ void __launch_bounds__(256)
k_scores(const float* __restrict__ uSum, const float* __restrict__ iSum,
         const int* __restrict__ uid, const int* __restrict__ iid,
         float* __restrict__ out, int nU, int nI)
{
    const int gw   = (blockIdx.x * 256 + threadIdx.x) >> 5;
    const int lane = threadIdx.x & 31;
    const int hi = lane >> 4, lo = lane & 15;
    const int tilesI = nI >> 4;
    const int tilesU = nU >> 4;
    const int ut = gw / tilesI, it = gw % tilesI;
    if (ut >= tilesU) return;

    const long ur = (long)(uid[ut * 16 + lo] - KN_ENT);
    const long ir = (long)iid[it * 16 + lo];

    v8f acc = {};
    #pragma unroll
    for (int kc = 0; kc < 2; ++kc) {
        V16U a, b;
        a.h[0] = cvt8(&uSum[ur * KD + kc * 32 + hi * 8]);
        a.h[1] = cvt8(&uSum[ur * KD + kc * 32 + 16 + hi * 8]);
        b.h[0] = cvt8(&iSum[ir * KD + kc * 32 + hi * 8]);
        b.h[1] = cvt8(&iSum[ir * KD + kc * 32 + 16 + hi * 8]);
        acc = __builtin_amdgcn_wmma_f32_16x16x32_f16(false, a.v, false, b.v,
                                                     (short)0, acc, false, false);
    }
    #pragma unroll
    for (int j = 0; j < 8; ++j) {
        out[(long)(ut * 16 + j + hi * 8) * nI + it * 16 + lo] = acc[j];
    }
}

// ---------------------------------------------------------------------------
// launch
// ---------------------------------------------------------------------------
extern "C" void kernel_launch(void* const* d_in, const int* in_sizes, int n_in,
                              void* d_out, int out_size, void* d_ws, size_t ws_size,
                              hipStream_t stream) {
    const float* emb     = (const float*)d_in[0];
    const float* rel     = (const float*)d_in[1];
    const float* Wk_w    = (const float*)d_in[2];
    const float* Wk_b    = (const float*)d_in[3];
    const float* Wa_w    = (const float*)d_in[4];
    const float* Wb_w    = (const float*)d_in[5];
    const int*   kg_h    = (const int*)d_in[6];
    const int*   kg_t    = (const int*)d_in[7];
    const int*   kg_r    = (const int*)d_in[8];
    const int*   ain_row = (const int*)d_in[9];
    const int*   ain_col = (const int*)d_in[10];
    const float* ain_val = (const float*)d_in[11];
    const int*   uid     = (const int*)d_in[12];
    const int*   iid     = (const int*)d_in[13];

    const int eKG = in_sizes[6];
    const int eCF = in_sizes[9];
    const int nU  = in_sizes[12];
    const int nI  = in_sizes[13];

    char* w = (char*)d_ws;
    size_t off = 0;
    auto take = [&](size_t bytes) -> float* {
        float* p = (float*)(w + off);
        off = (off + bytes + 255) & ~(size_t)255;
        return p;
    };
    float* attn      = take((size_t)eKG * 4);
    float* rmax      = take((size_t)KN_ENT * 4);
    float* rsum      = take((size_t)KN_ENT * 4);
    float* ent_cur   = take((size_t)KN_ENT * KD * 4);
    float* dual      = take((size_t)KN_ENT * KD * 4);
    float* kg_out    = take((size_t)KN_ENT * KD * 4);
    float* ain_out   = take((size_t)(KN_ENT + KN_USR) * KD * 4);
    float* users_cur = take((size_t)KN_USR * KD * 4);
    float* item_sum  = take((size_t)KN_ENT * KD * 4);
    float* user_sum  = take((size_t)KN_USR * KD * 4);
    (void)ws_size; (void)n_in; (void)out_size;

    const long entEl = (long)KN_ENT * KD;
    const long usrEl = (long)KN_USR * KD;

    hipMemcpyAsync(ent_cur,  emb, entEl * 4, hipMemcpyDeviceToDevice, stream);
    hipMemcpyAsync(dual,     emb, entEl * 4, hipMemcpyDeviceToDevice, stream);
    hipMemcpyAsync(item_sum, emb, entEl * 4, hipMemcpyDeviceToDevice, stream);
    hipMemcpyAsync(users_cur, emb + entEl, usrEl * 4, hipMemcpyDeviceToDevice, stream);
    hipMemcpyAsync(user_sum,  emb + entEl, usrEl * 4, hipMemcpyDeviceToDevice, stream);
    k_fill<<<(KN_ENT + 255) / 256, 256, 0, stream>>>(rmax, -INFINITY, (long)KN_ENT);
    k_fill<<<(KN_ENT + 255) / 256, 256, 0, stream>>>(rsum, 0.0f, (long)KN_ENT);

    k_logits<<<(eKG + 127) / 128, 256, 0, stream>>>(emb, rel, Wk_w, Wk_b,
                                                    kg_h, kg_t, kg_r, attn, rmax, eKG);
    k_exp <<<(eKG + 255) / 256, 256, 0, stream>>>(attn, kg_h, rmax, rsum, eKG);
    k_norm<<<(eKG + 255) / 256, 256, 0, stream>>>(attn, kg_h, rsum, eKG);

    const int spmmKGblocks = (int)(((long)eKG * 16 + 255) / 256);
    const int spmmCFblocks = (int)(((long)eCF * 16 + 255) / 256);
    const int fusionBlocks = ((KN_ENT + 15) / 16 + 7) / 8;
    for (int layer = 0; layer < 3; ++layer) {
        k_fill<<<(int)((entEl + 255) / 256), 256, 0, stream>>>(kg_out, 0.0f, entEl);
        k_spmm_kg<<<spmmKGblocks, 256, 0, stream>>>(attn, kg_h, kg_t, ent_cur, kg_out, eKG);

        k_fill<<<(int)((entEl + usrEl + 255) / 256), 256, 0, stream>>>(ain_out, 0.0f, entEl + usrEl);
        k_spmm_ain<<<spmmCFblocks, 256, 0, stream>>>(ain_val, ain_row, ain_col,
                                                     dual, users_cur, ain_out, eCF);

        k_fusion<<<fusionBlocks, 256, 0, stream>>>(kg_out, ain_out, Wa_w, Wb_w, dual, KN_ENT);

        k_add<<<(int)((entEl + 255) / 256), 256, 0, stream>>>(item_sum, ain_out, entEl);
        k_add<<<(int)((usrEl + 255) / 256), 256, 0, stream>>>(user_sum, ain_out + entEl, usrEl);

        hipMemcpyAsync(users_cur, ain_out + entEl, usrEl * 4, hipMemcpyDeviceToDevice, stream);
        float* tmp = ent_cur; ent_cur = kg_out; kg_out = tmp;
    }

    const int tilesU = nU >> 4, tilesI = nI >> 4;
    const int scoreBlocks = (tilesU * tilesI * 32 + 255) / 256;
    k_scores<<<scoreBlocks, 256, 0, stream>>>(user_sum, item_sum, uid, iid,
                                              (float*)d_out, nU, nI);
}
